// MultiHeadSelfAttention_67061619359963
// MI455X (gfx1250) — compile-verified, matrix operand order checked by reading, not run on hardware
//
#include <hip/hip_runtime.h>
#include <hip/hip_bf16.h>
#include <math.h>

#define H_DIM    2048
#define N_HEADS  16
#define HEAD_DIM 128
#define BATCH    2
#define SEQ      2048

typedef __attribute__((ext_vector_type(16))) __bf16 v16bf;
typedef __attribute__((ext_vector_type(8)))  float  v8f;

union Frag16 { v16bf v; uint4 q[2]; unsigned short s[16]; };
union Frag8f { v8f v; float f[8]; };

__device__ __forceinline__ unsigned short f2bf(float x) {
  unsigned int u = __float_as_uint(x);
  u += 0x7FFFu + ((u >> 16) & 1u);     // round-to-nearest-even
  return (unsigned short)(u >> 16);
}

__global__ void cvt_f32_bf16(const float* __restrict__ in,
                             unsigned short* __restrict__ out, int n) {
  int i = blockIdx.x * blockDim.x + threadIdx.x;
  int stride = gridDim.x * blockDim.x;
  for (; i < n; i += stride) out[i] = f2bf(in[i]);
}

// Y[M x N] = A[M x K](bf16, row-major) * W[K x N](bf16, row-major) + bias(f32)
// outMode 0: bf16 scatter to [b][h][s][d] planes (for Q/K/V)
// outMode 1: f32 row-major (final projection)
// Block: 256 threads = 8 waves, covers 128 rows x 128 cols of Y.
// Wave tile: 32 rows x 64 cols = 8 accumulators, 8 WMMA per K-step.
// Double-buffered LDS W tiles so global loads overlap WMMA.
__global__ __launch_bounds__(256) void gemm_bf16_wmma(
    const unsigned short* __restrict__ A,
    const unsigned short* __restrict__ W,
    const float* __restrict__ bias,
    unsigned short* __restrict__ outB,
    float* __restrict__ outF,
    int M, int N, int K, int outMode)
{
  __shared__ __align__(16) unsigned short wt[2][128 * 32]; // W^T tile [colLocal][kLocal]

  const int lane  = threadIdx.x & 31;
  const int wid   = threadIdx.x >> 5;
  const int l15   = lane & 15;
  const int half8 = (lane >> 4) << 3;          // 0 for lanes 0-15, 8 for 16-31
  const int rowBase = blockIdx.y * 128 + (wid & 3) * 32;
  const int colBlk  = blockIdx.x * 128;
  const int colBase = colBlk + (wid >> 2) * 64;

  // cooperative W-tile loader mapping (32 k x 128 cols)
  const int wkk = threadIdx.x >> 3;            // 0..31
  const int wc0 = (threadIdx.x & 7) * 16;      // 0..112

  Frag8f acc[2][4];
#pragma unroll
  for (int a = 0; a < 2; ++a)
#pragma unroll
    for (int i = 0; i < 4; ++i)
#pragma unroll
      for (int r = 0; r < 8; ++r) acc[a][i].f[r] = 0.f;

  const int NT = K / 32;

  // prologue: stage tile 0
  uint4 w0, w1;
  {
    const unsigned short* p = W + (size_t)wkk * N + colBlk + wc0;
    w0 = *(const uint4*)(p);
    w1 = *(const uint4*)(p + 8);
    const unsigned short* s0 = (const unsigned short*)&w0;
    const unsigned short* s1 = (const unsigned short*)&w1;
#pragma unroll
    for (int j = 0; j < 8; ++j) wt[0][(wc0 + j) * 32 + wkk] = s0[j];
#pragma unroll
    for (int j = 0; j < 8; ++j) wt[0][(wc0 + 8 + j) * 32 + wkk] = s1[j];
  }
  __syncthreads();

  for (int it = 0; it < NT; ++it) {
    const int kb  = it * 32;
    const int buf = it & 1;

    // issue next tile's global loads early (overlap with WMMA below)
    if (it + 1 < NT) {
      const unsigned short* p = W + (size_t)(kb + 32 + wkk) * N + colBlk + wc0;
      w0 = *(const uint4*)(p);
      w1 = *(const uint4*)(p + 8);
      if (it + 2 < NT)
        __builtin_prefetch(W + (size_t)(kb + 64 + wkk) * N + colBlk + wc0, 0, 3);
    }

    // A fragments for the two 16-row halves of the wave tile
    Frag16 a0, a1;
    {
      const unsigned short* p = A + (size_t)(rowBase + l15) * K + kb + half8;
      a0.q[0] = *(const uint4*)(p);
      a0.q[1] = *(const uint4*)(p + 16);
      const unsigned short* p2 = p + (size_t)16 * K;
      a1.q[0] = *(const uint4*)(p2);
      a1.q[1] = *(const uint4*)(p2 + 16);
    }

#pragma unroll
    for (int nb = 0; nb < 4; ++nb) {
      Frag16 bf; // B-layout: lane col = l15; element halves: K {half8..+7, half8+16..+23}
      const unsigned short* p =
          &wt[buf][((wid >> 2) * 64 + nb * 16 + l15) * 32 + half8];
      bf.q[0] = *(const uint4*)(p);
      bf.q[1] = *(const uint4*)(p + 16);
      acc[0][nb].v = __builtin_amdgcn_wmma_f32_16x16x32_bf16(
          false, a0.v, false, bf.v, (short)0, acc[0][nb].v, false, false);
      acc[1][nb].v = __builtin_amdgcn_wmma_f32_16x16x32_bf16(
          false, a1.v, false, bf.v, (short)0, acc[1][nb].v, false, false);
    }

    // stage next tile into the alternate buffer
    if (it + 1 < NT) {
      const unsigned short* s0 = (const unsigned short*)&w0;
      const unsigned short* s1 = (const unsigned short*)&w1;
#pragma unroll
      for (int j = 0; j < 8; ++j) wt[buf ^ 1][(wc0 + j) * 32 + wkk] = s0[j];
#pragma unroll
      for (int j = 0; j < 8; ++j) wt[buf ^ 1][(wc0 + 8 + j) * 32 + wkk] = s1[j];
    }
    __syncthreads();
  }

#pragma unroll
  for (int a = 0; a < 2; ++a) {
#pragma unroll
    for (int r = 0; r < 8; ++r) {
      int m = rowBase + a * 16 + r + half8;     // C-layout row
#pragma unroll
      for (int nb = 0; nb < 4; ++nb) {
        int col = colBase + nb * 16 + l15;      // C-layout col
        float val = acc[a][nb].f[r] + bias[col];
        if (outMode == 0) {
          int b = m >> 11;                      // m / SEQ   (M == BATCH*SEQ)
          int s = m & (SEQ - 1);
          int h = col >> 7;                     // col / HEAD_DIM
          int d = col & (HEAD_DIM - 1);
          outB[(((size_t)(b * N_HEADS + h)) * SEQ + s) * HEAD_DIM + d] = f2bf(val);
        } else {
          outF[(size_t)m * N + col] = val;
        }
      }
    }
  }
}

// Flash attention: block = (b, h, 128-query chunk); 8 waves x 16 queries each.
// Keys streamed in tiles of 32 with double-buffered LDS (K row-major, V transposed).
__global__ __launch_bounds__(256) void attn_wmma(
    const unsigned short* __restrict__ Qh,
    const unsigned short* __restrict__ Kh,
    const unsigned short* __restrict__ Vh,
    const int* __restrict__ mask,
    unsigned short* __restrict__ Ctx)
{
  __shared__ __align__(16) unsigned short kt[2][32 * 128];   // [key][dim]
  __shared__ __align__(16) unsigned short vt[2][128 * 32];   // [dim][key]
  __shared__ __align__(16) unsigned short ps[8 * 16 * 32];   // per-wave P staging

  const int lane  = threadIdx.x & 31;
  const int wid   = threadIdx.x >> 5;
  const int l15   = lane & 15;
  const int half8 = (lane >> 4) << 3;

  const int h = blockIdx.y;
  const int b = blockIdx.z;
  const size_t plane = ((size_t)(b * N_HEADS + h)) * SEQ * HEAD_DIM;
  const int qBase = blockIdx.x * 128 + wid * 16;

  const float scale = 0.08838834764831845f;  // 1/sqrt(128)

  // cooperative K/V tile loader mapping (32 keys x 128 dims)
  const int ckey = threadIdx.x >> 3;           // 0..31
  const int cd0  = (threadIdx.x & 7) * 16;     // 0..112

  // Q fragments: 16 rows x 128 dims, kept resident (4 A-fragments)
  Frag16 qf[4];
#pragma unroll
  for (int c = 0; c < 4; ++c) {
    const unsigned short* p =
        Qh + plane + (size_t)(qBase + l15) * HEAD_DIM + c * 32 + half8;
    qf[c].q[0] = *(const uint4*)(p);
    qf[c].q[1] = *(const uint4*)(p + 16);
  }

  Frag8f o[8];
  float m8[8], l8[8];
#pragma unroll
  for (int c = 0; c < 8; ++c)
#pragma unroll
    for (int r = 0; r < 8; ++r) o[c].f[r] = 0.f;
#pragma unroll
  for (int r = 0; r < 8; ++r) { m8[r] = -INFINITY; l8[r] = 0.f; }

  const int NT = SEQ / 32;
  uint4 kr0, kr1, vr0, vr1;

  // prologue: stage key-tile 0
  {
    const unsigned short* kp = Kh + plane + (size_t)ckey * HEAD_DIM + cd0;
    kr0 = *(const uint4*)(kp);
    kr1 = *(const uint4*)(kp + 8);
    const unsigned short* vp = Vh + plane + (size_t)ckey * HEAD_DIM + cd0;
    vr0 = *(const uint4*)(vp);
    vr1 = *(const uint4*)(vp + 8);
    *(uint4*)&kt[0][ckey * 128 + cd0]     = kr0;
    *(uint4*)&kt[0][ckey * 128 + cd0 + 8] = kr1;
    const unsigned short* v0s = (const unsigned short*)&vr0;
    const unsigned short* v1s = (const unsigned short*)&vr1;
#pragma unroll
    for (int j = 0; j < 8; ++j) vt[0][(cd0 + j) * 32 + ckey] = v0s[j];
#pragma unroll
    for (int j = 0; j < 8; ++j) vt[0][(cd0 + 8 + j) * 32 + ckey] = v1s[j];
  }
  __syncthreads();

  for (int it = 0; it < NT; ++it) {
    const int kt0 = it * 32;
    const int buf = it & 1;

    // issue next tile's global loads early (overlap with WMMA below)
    if (it + 1 < NT) {
      const unsigned short* kp =
          Kh + plane + (size_t)(kt0 + 32 + ckey) * HEAD_DIM + cd0;
      kr0 = *(const uint4*)(kp);
      kr1 = *(const uint4*)(kp + 8);
      const unsigned short* vp =
          Vh + plane + (size_t)(kt0 + 32 + ckey) * HEAD_DIM + cd0;
      vr0 = *(const uint4*)(vp);
      vr1 = *(const uint4*)(vp + 8);
      if (it + 2 < NT) {
        __builtin_prefetch(Kh + plane + (size_t)(kt0 + 64 + ckey) * HEAD_DIM + cd0, 0, 3);
        __builtin_prefetch(Vh + plane + (size_t)(kt0 + 64 + ckey) * HEAD_DIM + cd0, 0, 3);
      }
    }

    // S0/S1: 16x16 score tiles for keys kt0..+15 and kt0+16..+31
    Frag8f s0, s1;
#pragma unroll
    for (int r = 0; r < 8; ++r) { s0.f[r] = 0.f; s1.f[r] = 0.f; }
#pragma unroll
    for (int c = 0; c < 4; ++c) {
      Frag16 b0, b1; // B = K^T chunk; lane col = key, element = dim
      const unsigned short* p0 = &kt[buf][l15 * 128 + c * 32 + half8];
      b0.q[0] = *(const uint4*)(p0);
      b0.q[1] = *(const uint4*)(p0 + 16);
      const unsigned short* p1 = &kt[buf][(16 + l15) * 128 + c * 32 + half8];
      b1.q[0] = *(const uint4*)(p1);
      b1.q[1] = *(const uint4*)(p1 + 16);
      s0.v = __builtin_amdgcn_wmma_f32_16x16x32_bf16(
          false, qf[c].v, false, b0.v, (short)0, s0.v, false, false);
      s1.v = __builtin_amdgcn_wmma_f32_16x16x32_bf16(
          false, qf[c].v, false, b1.v, (short)0, s1.v, false, false);
    }

    // online softmax across the 32 new keys
#pragma unroll
    for (int r = 0; r < 8; ++r) {
      int m = qBase + r + half8;
      const int* mp = mask + ((size_t)b * SEQ + m) * SEQ + kt0;
      float sc0 = s0.f[r] * scale;
      float sc1 = s1.f[r] * scale;
      if (mp[l15] == 0)      sc0 = -INFINITY;
      if (mp[16 + l15] == 0) sc1 = -INFINITY;
      float rmax = fmaxf(sc0, sc1);
#pragma unroll
      for (int off = 1; off < 16; off <<= 1)
        rmax = fmaxf(rmax, __shfl_xor(rmax, off, 32)); // stays within 16-lane half
      float mnew  = fmaxf(m8[r], rmax);
      float alpha = __expf(m8[r] - mnew);
      float p0 = __expf(sc0 - mnew);
      float p1 = __expf(sc1 - mnew);
      float rsum = p0 + p1;
#pragma unroll
      for (int off = 1; off < 16; off <<= 1)
        rsum += __shfl_xor(rsum, off, 32);
      l8[r] = l8[r] * alpha + rsum;
      m8[r] = mnew;
#pragma unroll
      for (int c = 0; c < 8; ++c) o[c].f[r] *= alpha;
      // stage probabilities (C layout -> LDS row-major 16x32)
      ps[wid * 512 + (r + half8) * 32 + l15]      = f2bf(p0);
      ps[wid * 512 + (r + half8) * 32 + 16 + l15] = f2bf(p1);
    }

    // same-wave LDS RAW ordering before fragment reload
    asm volatile("s_wait_dscnt 0" ::: "memory");

    Frag16 pf; // P as A-fragment: 16 rows x 32 keys
    {
      const unsigned short* p = &ps[wid * 512 + l15 * 32 + half8];
      pf.q[0] = *(const uint4*)(p);
      pf.q[1] = *(const uint4*)(p + 16);
    }

#pragma unroll
    for (int c = 0; c < 8; ++c) {
      Frag16 vb; // B = V chunk (32 keys x 16 dims), read from transposed tile
      const unsigned short* p = &vt[buf][(c * 16 + l15) * 32 + half8];
      vb.q[0] = *(const uint4*)(p);
      vb.q[1] = *(const uint4*)(p + 16);
      o[c].v = __builtin_amdgcn_wmma_f32_16x16x32_bf16(
          false, pf.v, false, vb.v, (short)0, o[c].v, false, false);
    }

    // stage next tile into the alternate buffer
    if (it + 1 < NT) {
      *(uint4*)&kt[buf ^ 1][ckey * 128 + cd0]     = kr0;
      *(uint4*)&kt[buf ^ 1][ckey * 128 + cd0 + 8] = kr1;
      const unsigned short* v0s = (const unsigned short*)&vr0;
      const unsigned short* v1s = (const unsigned short*)&vr1;
#pragma unroll
      for (int j = 0; j < 8; ++j) vt[buf ^ 1][(cd0 + j) * 32 + ckey] = v0s[j];
#pragma unroll
      for (int j = 0; j < 8; ++j) vt[buf ^ 1][(cd0 + 8 + j) * 32 + ckey] = v1s[j];
    }
    __syncthreads();
  }

  // normalize + write context in [b][s][h*HD+d] (bf16) for the final GEMM
#pragma unroll
  for (int r = 0; r < 8; ++r) {
    int m = qBase + r + half8;
    float inv = 1.0f / l8[r];
#pragma unroll
    for (int c = 0; c < 8; ++c) {
      float val = o[c].f[r] * inv;
      Ctx[((size_t)(b * SEQ + m)) * H_DIM + h * HEAD_DIM + c * 16 + l15] = f2bf(val);
    }
  }
}

extern "C" void kernel_launch(void* const* d_in, const int* in_sizes, int n_in,
                              void* d_out, int out_size, void* d_ws, size_t ws_size,
                              hipStream_t stream)
{
  const float* X    = (const float*)d_in[0];
  const int*   mask = (const int*)d_in[1];
  const float* Wq   = (const float*)d_in[2];
  const float* bq   = (const float*)d_in[3];
  const float* Wk   = (const float*)d_in[4];
  const float* bk   = (const float*)d_in[5];
  const float* Wv   = (const float*)d_in[6];
  const float* bv   = (const float*)d_in[7];
  const float* Wo   = (const float*)d_in[8];
  const float* bo   = (const float*)d_in[9];
  float* out = (float*)d_out;

  const size_t nX = (size_t)BATCH * SEQ * H_DIM;  // 8M elements
  const size_t nW = (size_t)H_DIM * H_DIM;        // 4M elements

  unsigned short* ws = (unsigned short*)d_ws;
  unsigned short* Xh  = ws; ws += nX;
  unsigned short* Wqh = ws; ws += nW;
  unsigned short* Wkh = ws; ws += nW;
  unsigned short* Wvh = ws; ws += nW;
  unsigned short* Woh = ws; ws += nW;
  unsigned short* Qh  = ws; ws += nX;
  unsigned short* Kh  = ws; ws += nX;
  unsigned short* Vh  = ws; ws += nX;
  unsigned short* Ch  = ws; ws += nX;

  dim3 cblk(256), cgrid(1024);
  cvt_f32_bf16<<<cgrid, cblk, 0, stream>>>(X,  Xh,  (int)nX);
  cvt_f32_bf16<<<cgrid, cblk, 0, stream>>>(Wq, Wqh, (int)nW);
  cvt_f32_bf16<<<cgrid, cblk, 0, stream>>>(Wk, Wkh, (int)nW);
  cvt_f32_bf16<<<cgrid, cblk, 0, stream>>>(Wv, Wvh, (int)nW);
  cvt_f32_bf16<<<cgrid, cblk, 0, stream>>>(Wo, Woh, (int)nW);

  dim3 gblk(256), ggrid(H_DIM / 128, (BATCH * SEQ) / 128);
  gemm_bf16_wmma<<<ggrid, gblk, 0, stream>>>(Xh, Wqh, bq, Qh, nullptr,
                                             BATCH * SEQ, H_DIM, H_DIM, 0);
  gemm_bf16_wmma<<<ggrid, gblk, 0, stream>>>(Xh, Wkh, bk, Kh, nullptr,
                                             BATCH * SEQ, H_DIM, H_DIM, 0);
  gemm_bf16_wmma<<<ggrid, gblk, 0, stream>>>(Xh, Wvh, bv, Vh, nullptr,
                                             BATCH * SEQ, H_DIM, H_DIM, 0);

  dim3 ablk(256), agrid(SEQ / 128, N_HEADS, BATCH);
  attn_wmma<<<agrid, ablk, 0, stream>>>(Qh, Kh, Vh, mask, Ch);

  gemm_bf16_wmma<<<ggrid, gblk, 0, stream>>>(Ch, Woh, bo, nullptr, out,
                                             BATCH * SEQ, H_DIM, H_DIM, 1);
}
